// SpatialOCR_Module_73400991089403
// MI455X (gfx1250) — matrix operand order repaired and checked by reading, not compiled, not run on hardware
//
#include <hip/hip_runtime.h>
#include <hip/hip_bf16.h>

// ---------------------------------------------------------------------------
// SpatialOCR on MI455X (gfx1250): bf16 WMMA GEMM pipeline, f32 accumulate.
// ---------------------------------------------------------------------------

typedef __bf16 bf16x16 __attribute__((ext_vector_type(16)));
typedef float f32x8 __attribute__((ext_vector_type(8)));
typedef int v4i32 __attribute__((vector_size(16)));

constexpr int PIX = 128 * 128;   // 16384 pixels
constexpr int NB  = 8;           // batch

// GEMM tile config: 256 threads = 8 waves (wave32). 4 o-waves x 2 p-waves,
// each wave owns 2x2 tiles of 16x16 -> block tile 128(o) x 64(p).
constexpr int BO = 128, BP = 64, KC = 32, KPAD = 40;  // KPAD*2B=80B: 16B aligned, conflict-free

// CDNA5 async global->LDS DMA path (ASYNCcnt), guarded so compile cannot break.
#if defined(__has_builtin)
# if __has_builtin(__builtin_amdgcn_global_load_async_to_lds_b128) && \
     __has_builtin(__builtin_amdgcn_s_wait_asynccnt)
#  define USE_ASYNC_LDS 1
# endif
#endif
#ifndef USE_ASYNC_LDS
# define USE_ASYNC_LDS 0
#endif

#if USE_ASYNC_LDS
// Builtin signature (from hipcc diagnostic): param0 = v4i32* in global AS,
// param1 = v4i32* in LDS AS, then imm offset + imm cpol.
#define ASYNC_COPY16(gp, lp)                                                   \
  __builtin_amdgcn_global_load_async_to_lds_b128(                              \
      (__attribute__((address_space(1))) v4i32*)(gp),                          \
      (__attribute__((address_space(3))) v4i32*)(lp), 0, 0)
#define ASYNC_WAIT() __builtin_amdgcn_s_wait_asynccnt(0)
#else
#define ASYNC_COPY16(gp, lp) (*(uint4*)(lp) = *(const uint4*)(gp))
#define ASYNC_WAIT() ((void)0)
#endif

// ---- WMMA operand loaders (CDNA5 16-bit 16x32 A / 32x16 B VGPR layouts) ----
// A (16 rows x 32 K): lane l<16 -> row l, K = {0..7,16..23}; l>=16 -> K={8..15,24..31}
__device__ inline bf16x16 ldA(const __hip_bfloat16* base, int lane, int pitch) {
  int r = lane & 15, hi = lane >> 4;
  const __hip_bfloat16* p = base + r * pitch + hi * 8;
  union { uint4 u[2]; bf16x16 v; } u;
  u.u[0] = *(const uint4*)(p);        // K = hi*8 .. +7
  u.u[1] = *(const uint4*)(p + 16);   // K = hi*8+16 .. +23
  return u.v;
}
// B (32 K x 16 cols), stored col-major-as-rows: lane l<16 -> col l, K = {0..7,16..23}; l>=16 -> K={8..15,24..31}
__device__ inline bf16x16 ldB(const __hip_bfloat16* base, int lane, int pitch) {
  int c = lane & 15, hi = lane >> 4;
  const __hip_bfloat16* p = base + c * pitch + hi * 8;
  union { uint4 u[2]; bf16x16 v; } u;
  u.u[0] = *(const uint4*)(p);        // K = hi*8 .. +7
  u.u[1] = *(const uint4*)(p + 16);   // K = hi*8+16 .. +23
  return u.v;
}

__device__ inline f32x8 wmma_bf16(bf16x16 a, bf16x16 b, f32x8 c) {
  return __builtin_amdgcn_wmma_f32_16x16x32_bf16(false, a, false, b, (short)0, c,
                                                 false, false);
}

// Pack two fp32 into one dword of {lo: bf16(lo), hi: bf16(hi)}.
__device__ inline unsigned pack2_bf16(float lo, float hi) {
  union { __hip_bfloat16 b; unsigned short u; } a{__float2bfloat16(lo)},
      c{__float2bfloat16(hi)};
  return (unsigned)a.u | ((unsigned)c.u << 16);
}

// ---------------------------------------------------------------------------
// Fused conv1x1 + BN-affine + ReLU GEMM:
//   out[n,o,p] = relu( s[o] * ( sum_k W[o,k]*X1[n,k,p]  (+ W[o,K1+k]*X2[n,k,p]) ) + b[o] )
// X2 (when present) is the fp32 feats stream -> implements concat for free.
// ---------------------------------------------------------------------------
template <bool IN1_F32, bool HAS2, bool OUT_F32>
__global__ __launch_bounds__(256) void cbr_gemm(
    const void* __restrict__ in1, const float* __restrict__ in2,
    const __hip_bfloat16* __restrict__ W, const float* __restrict__ s,
    const float* __restrict__ bias, void* __restrict__ out, int O, int K1, int K2) {
  const int n = blockIdx.z;
  const int o0 = blockIdx.y * BO;
  const int p0 = blockIdx.x * BP;
  const int Kt = K1 + (HAS2 ? K2 : 0);
  const int t = threadIdx.x;
  const int lane = t & 31, wv = t >> 5;
  const int wo = wv >> 1, wp = wv & 1;

  __shared__ __align__(16) __hip_bfloat16 sW[2][BO][KPAD];
  __shared__ __align__(16) __hip_bfloat16 sX[2][BP][KPAD];
  __shared__ float sS[BO], sB[BO];

  if (t < BO) { sS[t] = s[o0 + t]; sB[t] = bias[o0 + t]; }

  const __hip_bfloat16* in1b = (const __hip_bfloat16*)in1;
  const float* in1f = (const float*)in1;
  const size_t in1off = (size_t)n * K1 * PIX;
  const size_t in2off = HAS2 ? (size_t)n * K2 * PIX : 0;

  // 2x2 (k,p) quad staging indices (shared by all X variants).
  const int xp = (t & 31) * 2;   // p-pair base
  const int xk = t >> 5;         // k-pair base (0..7)

  auto stage = [&](int kg, int buf) {
    // ---- W tile: 128 rows x 32 k, 16B chunks; async DMA to LDS when available.
    int ci = t;
#pragma unroll
    for (int i = 0; i < 2; i++, ci += 256) {
      int row = ci >> 2, kof = (ci & 3) * 8;
      ASYNC_COPY16(W + (size_t)(o0 + row) * Kt + kg + kof, &sW[buf][row][kof]);
    }
    // ---- X tile: 32k x 64p from [k][p] global, transposed into LDS [p][k].
    // Each thread handles two 2x2 quads: b32/float2 row-pair loads, v_perm /
    // packed-cvt repack, b32 LDS stores.
    if constexpr (HAS2) {
      if (kg >= K1) {
        const float* s0 = in2 + in2off + (size_t)(kg - K1) * PIX + p0 + xp;
#pragma unroll
        for (int j = 0; j < 2; j++) {
          int k = (xk + j * 8) * 2;
          float2 fa = *(const float2*)(s0 + (size_t)k * PIX);
          float2 fb = *(const float2*)(s0 + (size_t)(k + 1) * PIX);
          *(unsigned*)&sX[buf][xp][k] = pack2_bf16(fa.x, fb.x);
          *(unsigned*)&sX[buf][xp + 1][k] = pack2_bf16(fa.y, fb.y);
        }
        return;
      }
    }
    if constexpr (IN1_F32) {
      const float* s0 = in1f + in1off + (size_t)kg * PIX + p0 + xp;
#pragma unroll
      for (int j = 0; j < 2; j++) {
        int k = (xk + j * 8) * 2;
        float2 fa = *(const float2*)(s0 + (size_t)k * PIX);
        float2 fb = *(const float2*)(s0 + (size_t)(k + 1) * PIX);
        *(unsigned*)&sX[buf][xp][k] = pack2_bf16(fa.x, fb.x);
        *(unsigned*)&sX[buf][xp + 1][k] = pack2_bf16(fa.y, fb.y);
      }
      if (kg + 2 * KC < Kt)
        __builtin_prefetch(s0 + (size_t)2 * KC * PIX, 0, 3);  // global_prefetch_b8
    } else {
      const __hip_bfloat16* s0 = in1b + in1off + (size_t)kg * PIX + p0 + xp;
#pragma unroll
      for (int j = 0; j < 2; j++) {
        int k = (xk + j * 8) * 2;
        unsigned lo = *(const unsigned*)(s0 + (size_t)k * PIX);
        unsigned hi = *(const unsigned*)(s0 + (size_t)(k + 1) * PIX);
        *(unsigned*)&sX[buf][xp][k] = __builtin_amdgcn_perm(hi, lo, 0x05040100u);
        *(unsigned*)&sX[buf][xp + 1][k] =
            __builtin_amdgcn_perm(hi, lo, 0x07060302u);
      }
      if (kg + 2 * KC < K1)
        __builtin_prefetch(s0 + (size_t)2 * KC * PIX, 0, 3);
    }
  };

  f32x8 acc[2][2] = {};

  stage(0, 0);
  ASYNC_WAIT();
  __syncthreads();
  for (int kg = 0; kg < Kt; kg += KC) {
    int cur = (kg / KC) & 1;
    if (kg + KC < Kt) stage(kg + KC, cur ^ 1);

    bf16x16 a0 = ldA(&sW[cur][wo * 32 + 0][0], lane, KPAD);
    bf16x16 a1 = ldA(&sW[cur][wo * 32 + 16][0], lane, KPAD);
    bf16x16 b0 = ldB(&sX[cur][wp * 32 + 0][0], lane, KPAD);
    bf16x16 b1 = ldB(&sX[cur][wp * 32 + 16][0], lane, KPAD);
    acc[0][0] = wmma_bf16(a0, b0, acc[0][0]);
    acc[0][1] = wmma_bf16(a0, b1, acc[0][1]);
    acc[1][0] = wmma_bf16(a1, b0, acc[1][0]);
    acc[1][1] = wmma_bf16(a1, b1, acc[1][1]);
    ASYNC_WAIT();
    __syncthreads();
  }

  const size_t outoff = (size_t)n * O * PIX;
#pragma unroll
  for (int to = 0; to < 2; to++)
#pragma unroll
    for (int tp = 0; tp < 2; tp++)
#pragma unroll
      for (int g = 0; g < 8; g++) {
        int rl = wo * 32 + to * 16 + (lane >> 4) * 8 + g;   // local o
        int pp = p0 + wp * 32 + tp * 16 + (lane & 15);      // global p
        float y = acc[to][tp][g] * sS[rl] + sB[rl];
        y = fmaxf(y, 0.f);
        if constexpr (OUT_F32)
          ((float*)out)[outoff + (size_t)(o0 + rl) * PIX + pp] = y;
        else
          ((__hip_bfloat16*)out)[outoff + (size_t)(o0 + rl) * PIX + pp] =
              __float2bfloat16(y);
      }
}

// ---------------------------------------------------------------------------
// Fused attention: simT = keyT(64x256) x qtT(256x64p), softmax over m=64 in
// LDS, ctx = value(256x64) x attT(64x64p). sim/att never hit HBM.
// ---------------------------------------------------------------------------
constexpr int QPITCH = 264;  // 256+8: 528B rows (16B aligned), conflict-free
constexpr int APITCH = 72;   // 144B rows (16B aligned)

__global__ __launch_bounds__(256) void attn_kernel(
    const __hip_bfloat16* __restrict__ qt,    // [N][256][P]
    const __hip_bfloat16* __restrict__ keyT,  // [N][64][256]
    const __hip_bfloat16* __restrict__ val,   // [N][256][64]
    __hip_bfloat16* __restrict__ ctx)         // [N][256][P]
{
  const int n = blockIdx.y;
  const int p0 = blockIdx.x * 64;
  const int t = threadIdx.x, lane = t & 31, wv = t >> 5;

  __shared__ __align__(16) __hip_bfloat16 sQ[64][QPITCH];  // [p][k]
  __shared__ float sSim[64][65];                           // [p][m]
  __shared__ __align__(16) __hip_bfloat16 sAtt[64][APITCH];// [p][m]

  // Stage qt tile transposed (64p x 256k) with 2x2 quad packing.
  {
    int p = (t & 31) * 2, kk0 = t >> 5;
    const __hip_bfloat16* q = qt + (size_t)n * 256 * PIX + p0 + p;
#pragma unroll 4
    for (int j = 0; j < 16; j++) {
      int k = (kk0 + j * 8) * 2;
      unsigned lo = *(const unsigned*)(q + (size_t)k * PIX);
      unsigned hi = *(const unsigned*)(q + (size_t)(k + 1) * PIX);
      *(unsigned*)&sQ[p][k] = __builtin_amdgcn_perm(hi, lo, 0x05040100u);
      *(unsigned*)&sQ[p + 1][k] = __builtin_amdgcn_perm(hi, lo, 0x07060302u);
    }
  }
  __syncthreads();

  // simT tiles: M = region index m (64), N = pixels. A = keyT from global (L2-hot).
  const __hip_bfloat16* kT = keyT + (size_t)n * 64 * 256;
  {
    int pt = wv & 3, mh = wv >> 2;
    f32x8 acc[2] = {};
    for (int kc = 0; kc < 256; kc += 32) {
      bf16x16 b = ldB(&sQ[pt * 16][kc], lane, QPITCH);
      bf16x16 a0 = ldA(kT + (size_t)(mh * 32 + 0) * 256 + kc, lane, 256);
      bf16x16 a1 = ldA(kT + (size_t)(mh * 32 + 16) * 256 + kc, lane, 256);
      acc[0] = wmma_bf16(a0, b, acc[0]);
      acc[1] = wmma_bf16(a1, b, acc[1]);
    }
    int pc = pt * 16 + (lane & 15);
#pragma unroll
    for (int tm = 0; tm < 2; tm++)
#pragma unroll
      for (int g = 0; g < 8; g++) {
        int m = mh * 32 + tm * 16 + (lane >> 4) * 8 + g;
        sSim[pc][m] = acc[tm][g] * 0.0625f;  // KEY_CH^-0.5 = 1/16
      }
  }
  __syncthreads();

  // Softmax over m=64 per pixel (one thread per pixel; tiny).
  if (t < 64) {
    float mx = -1e30f;
#pragma unroll 8
    for (int m = 0; m < 64; m++) mx = fmaxf(mx, sSim[t][m]);
    float sum = 0.f;
#pragma unroll 8
    for (int m = 0; m < 64; m++) sum += __expf(sSim[t][m] - mx);
    float inv = 1.f / sum;
#pragma unroll 8
    for (int m = 0; m < 64; m++)
      sAtt[t][m] = __float2bfloat16(__expf(sSim[t][m] - mx) * inv);
  }
  __syncthreads();

  // ctx[kc,p] = sum_m value[kc,m] * att[p,m] : M-tiles = 16 (kc=256), K = 64.
  const __hip_bfloat16* V = val + (size_t)n * 256 * 64;
  __hip_bfloat16* C = ctx + (size_t)n * 256 * PIX;
  int pt = wv & 3, half = wv >> 2;
  for (int mt8 = 0; mt8 < 8; mt8++) {
    int mtile = half * 8 + mt8;
    f32x8 acc = {};
#pragma unroll
    for (int ks = 0; ks < 2; ks++) {
      bf16x16 a = ldA(V + (size_t)(mtile * 16) * 64 + ks * 32, lane, 64);
      bf16x16 b = ldB(&sAtt[pt * 16][ks * 32], lane, APITCH);
      acc = wmma_bf16(a, b, acc);
    }
    int pc = p0 + pt * 16 + (lane & 15);
#pragma unroll
    for (int g = 0; g < 8; g++) {
      int kc = mtile * 16 + (lane >> 4) * 8 + g;
      C[(size_t)kc * PIX + pc] = __float2bfloat16(acc[g]);
    }
  }
}

// ---------------------------------------------------------------------------
// Proxy-side (M=64) stages: negligible FLOPs, scalar f32.
// ---------------------------------------------------------------------------
__global__ __launch_bounds__(256) void proxy_stage1(
    const float* __restrict__ pr, const float* __restrict__ wo1,
    const float* __restrict__ so1, const float* __restrict__ bo1,
    const float* __restrict__ wd, const float* __restrict__ sd,
    const float* __restrict__ bd, float* __restrict__ k1,
    __hip_bfloat16* __restrict__ val) {
  int n = blockIdx.y, which = blockIdx.z;
  int o = threadIdx.x;
  int m0 = blockIdx.x * 8;
  const float* w = which ? wd : wo1;
  const float* ss = which ? sd : so1;
  const float* bb = which ? bd : bo1;
  const float* x = pr + (size_t)n * 512 * 64;
  for (int mi = 0; mi < 8; mi++) {
    int m = m0 + mi;
    float acc = 0.f;
    for (int c = 0; c < 512; c++) acc += w[o * 512 + c] * x[c * 64 + m];
    float y = fmaxf(acc * ss[o] + bb[o], 0.f);
    if (which) val[((size_t)n * 256 + o) * 64 + m] = __float2bfloat16(y);
    else       k1[((size_t)n * 256 + o) * 64 + m] = y;
  }
}

__global__ __launch_bounds__(256) void proxy_stage2(
    const float* __restrict__ k1, const float* __restrict__ wo2,
    const float* __restrict__ so2, const float* __restrict__ bo2,
    __hip_bfloat16* __restrict__ keyT)  // [N][64][256] (transposed!)
{
  int n = blockIdx.y;
  int o = threadIdx.x;
  int m0 = blockIdx.x * 8;
  const float* x = k1 + (size_t)n * 256 * 64;
  for (int mi = 0; mi < 8; mi++) {
    int m = m0 + mi;
    float acc = 0.f;
    for (int c = 0; c < 256; c++) acc += wo2[o * 256 + c] * x[c * 64 + m];
    float y = fmaxf(acc * so2[o] + bo2[o], 0.f);
    keyT[((size_t)n * 64 + m) * 256 + o] = __float2bfloat16(y);
  }
}

__global__ void f32_to_bf16(const float* __restrict__ src,
                            __hip_bfloat16* __restrict__ dst, int n) {
  int i = blockIdx.x * blockDim.x + threadIdx.x;
  if (i < n) dst[i] = __float2bfloat16(src[i]);
}

// ---------------------------------------------------------------------------
extern "C" void kernel_launch(void* const* d_in, const int* in_sizes, int n_in,
                              void* d_out, int out_size, void* d_ws,
                              size_t ws_size, hipStream_t stream) {
  (void)in_sizes; (void)n_in; (void)out_size; (void)ws_size;
  const float* feats = (const float*)d_in[0];   // [8,512,128,128]
  const float* proxy = (const float*)d_in[1];   // [8,512,64,1]
  const float* wp1 = (const float*)d_in[2];
  const float* sp1 = (const float*)d_in[3];
  const float* bp1 = (const float*)d_in[4];
  const float* wp2 = (const float*)d_in[5];
  const float* sp2 = (const float*)d_in[6];
  const float* bp2 = (const float*)d_in[7];
  const float* wo1 = (const float*)d_in[8];
  const float* so1 = (const float*)d_in[9];
  const float* bo1 = (const float*)d_in[10];
  const float* wo2 = (const float*)d_in[11];
  const float* so2 = (const float*)d_in[12];
  const float* bo2 = (const float*)d_in[13];
  const float* wd = (const float*)d_in[14];
  const float* sd = (const float*)d_in[15];
  const float* bd = (const float*)d_in[16];
  const float* wu = (const float*)d_in[17];
  const float* su = (const float*)d_in[18];
  const float* bu = (const float*)d_in[19];
  const float* wc = (const float*)d_in[20];
  const float* sc = (const float*)d_in[21];
  const float* bc = (const float*)d_in[22];

  // Workspace layout (256B aligned slabs).
  char* ws = (char*)d_ws;
  size_t off = 0;
  auto alloc = [&](size_t bytes) {
    size_t o = off;
    off += (bytes + 255) & ~(size_t)255;
    return o;
  };
  const size_t nWP1 = 256 * 512, nWP2 = 256 * 256, nWU = 512 * 256,
               nWC = 512 * 1024;
  __hip_bfloat16* wp1b = (__hip_bfloat16*)(ws + alloc(nWP1 * 2));
  __hip_bfloat16* wp2b = (__hip_bfloat16*)(ws + alloc(nWP2 * 2));
  __hip_bfloat16* wub  = (__hip_bfloat16*)(ws + alloc(nWU * 2));
  __hip_bfloat16* wcb  = (__hip_bfloat16*)(ws + alloc(nWC * 2));
  float* k1f           = (float*)(ws + alloc((size_t)NB * 256 * 64 * 4));
  __hip_bfloat16* valb = (__hip_bfloat16*)(ws + alloc((size_t)NB * 256 * 64 * 2));
  __hip_bfloat16* keyTb= (__hip_bfloat16*)(ws + alloc((size_t)NB * 64 * 256 * 2));
  // Big slabs: A = t1 then ctx; B..C = qt then ctx2 (ctx2 spans B+C).
  const size_t slab = (size_t)NB * 256 * PIX * 2;  // 67 MB
  __hip_bfloat16* bufA = (__hip_bfloat16*)(ws + alloc(slab));
  __hip_bfloat16* bufB = (__hip_bfloat16*)(ws + alloc(slab));
  (void)alloc(slab);  // bufC, contiguous after bufB -> ctx2 = bufB (134 MB)

  // 1) Weight conversion fp32 -> bf16.
  f32_to_bf16<<<(int)((nWP1 + 255) / 256), 256, 0, stream>>>(wp1, wp1b, (int)nWP1);
  f32_to_bf16<<<(int)((nWP2 + 255) / 256), 256, 0, stream>>>(wp2, wp2b, (int)nWP2);
  f32_to_bf16<<<(int)((nWU + 255) / 256), 256, 0, stream>>>(wu, wub, (int)nWU);
  f32_to_bf16<<<(int)((nWC + 255) / 256), 256, 0, stream>>>(wc, wcb, (int)nWC);

  // 2) Proxy-side key/value (tiny).
  proxy_stage1<<<dim3(8, NB, 2), 256, 0, stream>>>(proxy, wo1, so1, bo1, wd, sd,
                                                   bd, k1f, valb);
  proxy_stage2<<<dim3(8, NB), 256, 0, stream>>>(k1f, wo2, so2, bo2, keyTb);

  // 3) t1 = cbr(x, wp1)          [N,256,P]  (x is f32)
  cbr_gemm<true, false, false><<<dim3(PIX / BP, 256 / BO, NB), 256, 0, stream>>>(
      feats, nullptr, wp1b, sp1, bp1, bufA, 256, 512, 0);
  // 4) qt = cbr(t1, wp2)         [N,256,P]
  cbr_gemm<false, false, false><<<dim3(PIX / BP, 256 / BO, NB), 256, 0, stream>>>(
      bufA, nullptr, wp2b, sp2, bp2, bufB, 256, 256, 0);
  // 5) attention: ctx = softmax(qt^T key / 16) @ v^T   [N,256,P] (overwrites t1 slab)
  attn_kernel<<<dim3(PIX / 64, NB), 256, 0, stream>>>(bufB, keyTb, valb, bufA);
  // 6) ctx2 = cbr(ctx, wu)       [N,512,P] (overwrites qt slab + spare)
  cbr_gemm<false, false, false><<<dim3(PIX / BP, 512 / BO, NB), 256, 0, stream>>>(
      bufA, nullptr, wub, su, bu, bufB, 512, 256, 0);
  // 7) out = cbr([ctx2; x], wc)  [N,512,P] f32 — concat fused as K=512+512
  cbr_gemm<false, true, true><<<dim3(PIX / BP, 512 / BO, NB), 256, 0, stream>>>(
      bufB, feats, wcb, sc, bc, d_out, 512, 512, 512);
}
